// GraspSamplerDecoder_67577015435424
// MI455X (gfx1250) — matrix operand order repaired and checked by reading, not compiled, not run on hardware
//
#include <hip/hip_runtime.h>
#include <hip/hip_bf16.h>
#include <cstdint>

typedef __attribute__((ext_vector_type(16))) _Float16 v16h;
typedef __attribute__((ext_vector_type(8)))  _Float16 v8h;
typedef __attribute__((ext_vector_type(8)))  float    v8f;
typedef __attribute__((ext_vector_type(4)))  unsigned int v4u;
typedef __attribute__((ext_vector_type(4)))  int v4i;
typedef __attribute__((ext_vector_type(8)))  int v8i;

// ---------------------------------------------------------------------------
// TDM: stage one 64(N) x 32(K) f16 weight tile into LDS.
// D# group0: count=1 | lds_addr | global_addr | type=2.
// D# group1: data_size=2B, tensor_dim0=32, tensor_dim1=64, tile_dim0=32,
//            tile_dim1=64, tensor_dim0_stride=Cp (elements).
// ---------------------------------------------------------------------------
__device__ __forceinline__ void tdm_load_b_tile(const _Float16* g, unsigned lds_byte_off, int Cp) {
  unsigned long long ga = (unsigned long long)(uintptr_t)g;
  v4u g0;
  g0.x = 1u;                                               // count=1
  g0.y = lds_byte_off;                                     // lds_addr (bits 63:32)
  g0.z = (unsigned)(ga & 0xffffffffull);                   // global_addr lo
  g0.w = (unsigned)((ga >> 32) & 0x01ffffffull) | 0x80000000u; // addr hi | type=2
  v8i g1;
  g1[0] = 0x00010000;                                      // data_size=1 (2 bytes)
  g1[1] = (int)(32u << 16);                                // tensor_dim0[15:0] @63:48
  g1[2] = (int)(64u << 16);                                // tensor_dim1[15:0] @95:80 (dim0 hi=0)
  g1[3] = (int)(32u << 16);                                // tile_dim0 @127:112 (dim1 hi=0)
  g1[4] = 64;                                              // tile_dim1=64, tile_dim2=0
  g1[5] = Cp;                                              // tensor_dim0_stride lo
  g1[6] = 0;
  g1[7] = 0;
  v4i z4 = {0, 0, 0, 0};
#if defined(__clang_major__) && (__clang_major__ >= 23)
  v8i z8 = {0, 0, 0, 0, 0, 0, 0, 0};
  __builtin_amdgcn_tensor_load_to_lds(g0, g1, z4, z4, z8, 0);
#else
  __builtin_amdgcn_tensor_load_to_lds(g0, g1, z4, z4, 0);
#endif
}

// ---------------------------------------------------------------------------
// Weight convert: f32 [O,C] -> f16 [O,Cp] with zero pad (Cp multiple of 32)
// ---------------------------------------------------------------------------
__global__ void k_convert_w(const float* __restrict__ src, _Float16* __restrict__ dst,
                            int O, int C, int Cp) {
  int i = blockIdx.x * blockDim.x + threadIdx.x;
  if (i >= O * Cp) return;
  int o = i / Cp, c = i % Cp;
  dst[i] = (c < C) ? (_Float16)src[o * C + c] : (_Float16)0.f;
}

// ---------------------------------------------------------------------------
// Farthest point sampling: one block per cloud, cloud cached in LDS.
// ---------------------------------------------------------------------------
#define FPS_MAXN 2048
__global__ __launch_bounds__(256) void k_fps(const float* __restrict__ xyz,
                                             int N, int npoint,
                                             float* __restrict__ new_xyz) {
  int b = blockIdx.x;
  int t = threadIdx.x;
  const float* p = xyz + (size_t)b * N * 3;
  __shared__ float sx[FPS_MAXN], sy[FPS_MAXN], sz[FPS_MAXN], sd[FPS_MAXN];
  __shared__ float rv[256];
  __shared__ int   ri[256];
  __shared__ float fc[3];
  __shared__ int   sfar;
  for (int n = t; n < N; n += blockDim.x) {
    sx[n] = p[n * 3 + 0];
    sy[n] = p[n * 3 + 1];
    sz[n] = p[n * 3 + 2];
    sd[n] = 1e10f;
  }
  if (t == 0) sfar = 0;
  __syncthreads();
  for (int i = 0; i < npoint; ++i) {
    if (t == 0) {
      int far = sfar;
      float* o = new_xyz + ((size_t)b * npoint + i) * 3;
      o[0] = sx[far]; o[1] = sy[far]; o[2] = sz[far];
      fc[0] = sx[far]; fc[1] = sy[far]; fc[2] = sz[far];
    }
    __syncthreads();
    float bx = fc[0], by = fc[1], bz = fc[2];
    float bestv = -1.f; int besti = 0;
    for (int n = t; n < N; n += blockDim.x) {
      float dx = sx[n] - bx, dy = sy[n] - by, dz = sz[n] - bz;
      float d = dx * dx + dy * dy + dz * dz;
      float dm = fminf(sd[n], d);
      sd[n] = dm;
      if (dm > bestv) { bestv = dm; besti = n; }   // ascending n -> first max kept
    }
    rv[t] = bestv; ri[t] = besti;
    __syncthreads();
    for (int s = 128; s > 0; s >>= 1) {
      if (t < s) {
        if (rv[t + s] > rv[t] || (rv[t + s] == rv[t] && ri[t + s] < ri[t])) {
          rv[t] = rv[t + s]; ri[t] = ri[t + s];
        }
      }
      __syncthreads();
    }
    if (t == 0) sfar = ri[0];
    __syncthreads();
  }
}

// ---------------------------------------------------------------------------
// SA0 ball-query + grouping. One wave32 per (b,s).
// ---------------------------------------------------------------------------
__global__ __launch_bounds__(32) void k_group0(const float* __restrict__ pc,
                                               const float* __restrict__ z,
                                               const float* __restrict__ new_xyz,
                                               float radius2, int N, int S, int nsample,
                                               _Float16* __restrict__ Xo, int Cp) {
  int bs = blockIdx.x;
  int b  = bs / S;
  int lane = threadIdx.x;
  __shared__ int neigh[128];
  const float* p = pc + (size_t)b * N * 3;
  float cx = new_xyz[(size_t)bs * 3 + 0];
  float cy = new_xyz[(size_t)bs * 3 + 1];
  float cz = new_xyz[(size_t)bs * 3 + 2];
  int count = 0;
  for (int base = 0; base < N; base += 32) {
    int n = base + lane;
    bool ok = false;
    if (n < N) {
      float dx = p[n * 3 + 0] - cx, dy = p[n * 3 + 1] - cy, dz = p[n * 3 + 2] - cz;
      ok = (dx * dx + dy * dy + dz * dz) <= radius2;
    }
    unsigned msk = (unsigned)__ballot(ok);
    int slot = count + __popc(msk & ((1u << lane) - 1u));
    if (ok && slot < nsample) neigh[slot] = n;
    count += __popc(msk);
    if (count >= nsample) break;
  }
  if (count > nsample) count = nsample;
  __syncthreads();
  int first = neigh[0];
  for (int j = lane; j < nsample; j += 32)
    if (j >= count) neigh[j] = first;
  __syncthreads();
  float z0 = z[b * 2 + 0], z1 = z[b * 2 + 1];
  for (int j = lane; j < nsample; j += 32) {
    int id = neigh[j];
    _Float16* o = Xo + ((size_t)bs * nsample + j) * Cp;
    float px = p[id * 3 + 0], py = p[id * 3 + 1], pz = p[id * 3 + 2];
    o[0] = (_Float16)(px - cx); o[1] = (_Float16)(py - cy); o[2] = (_Float16)(pz - cz);
    o[3] = (_Float16)px;        o[4] = (_Float16)py;        o[5] = (_Float16)pz;
    o[6] = (_Float16)z0;        o[7] = (_Float16)z1;
    for (int c = 8; c < Cp; ++c) o[c] = (_Float16)0.f;
  }
}

// SA1 grouping: features [dxyz, points1(f16 x CF)] padded to Cp.
__global__ __launch_bounds__(32) void k_group1(const float* __restrict__ xyz,
                                               const _Float16* __restrict__ pts,
                                               const float* __restrict__ new_xyz,
                                               float radius2, int N, int S, int nsample, int CF,
                                               _Float16* __restrict__ Xo, int Cp) {
  int bs = blockIdx.x;
  int b  = bs / S;
  int lane = threadIdx.x;
  __shared__ int neigh[128];
  const float* p = xyz + (size_t)b * N * 3;
  float cx = new_xyz[(size_t)bs * 3 + 0];
  float cy = new_xyz[(size_t)bs * 3 + 1];
  float cz = new_xyz[(size_t)bs * 3 + 2];
  int count = 0;
  for (int base = 0; base < N; base += 32) {
    int n = base + lane;
    bool ok = false;
    if (n < N) {
      float dx = p[n * 3 + 0] - cx, dy = p[n * 3 + 1] - cy, dz = p[n * 3 + 2] - cz;
      ok = (dx * dx + dy * dy + dz * dz) <= radius2;
    }
    unsigned msk = (unsigned)__ballot(ok);
    int slot = count + __popc(msk & ((1u << lane) - 1u));
    if (ok && slot < nsample) neigh[slot] = n;
    count += __popc(msk);
    if (count >= nsample) break;
  }
  if (count > nsample) count = nsample;
  __syncthreads();
  int first = neigh[0];
  for (int j = lane; j < nsample; j += 32)
    if (j >= count) neigh[j] = first;
  __syncthreads();
  for (int j = lane; j < nsample; j += 32) {
    int id = neigh[j];
    _Float16* o = Xo + ((size_t)bs * nsample + j) * Cp;
    o[0] = (_Float16)(p[id * 3 + 0] - cx);
    o[1] = (_Float16)(p[id * 3 + 1] - cy);
    o[2] = (_Float16)(p[id * 3 + 2] - cz);
    const _Float16* f = pts + ((size_t)b * N + id) * CF;
    for (int c = 0; c < CF; ++c) o[3 + c] = f[c];
    for (int c = 3 + CF; c < Cp; ++c) o[c] = (_Float16)0.f;
  }
}

// SA2 (group_all): x = concat(xyz, points) padded to Cp.
__global__ void k_group_all(const float* __restrict__ xyz, const _Float16* __restrict__ pts,
                            _Float16* __restrict__ Xo, int B, int Npts, int C, int Cp) {
  int i = blockIdx.x * blockDim.x + threadIdx.x;
  int total = B * Npts * Cp;
  if (i >= total) return;
  int c = i % Cp;
  int row = i / Cp;
  _Float16 v = (_Float16)0.f;
  if (c < 3)            v = (_Float16)xyz[(size_t)row * 3 + c];
  else if (c < 3 + C)   v = pts[(size_t)row * C + (c - 3)];
  Xo[i] = v;
}

// ---------------------------------------------------------------------------
// WMMA GEMM: Y[R,O] = X[R,Cp] * W[O,Cp]^T + bias.  R%64==0, O%64==0, Cp%32==0.
// Block = 256 thr = 8 waves (4 in M x 2 in N); block tile 64x64.
// Wave tile 16(M) x 32(N): one A fragment feeds two WMMAs.
// B panel staged into LDS by the Tensor Data Mover (wave 0 issues), double
// buffered; A fragments loaded directly from global (b128).
// ---------------------------------------------------------------------------
__global__ __launch_bounds__(256) void k_gemm_wmma(const _Float16* __restrict__ X,
                                                   const _Float16* __restrict__ W,
                                                   const float* __restrict__ bias,
                                                   _Float16* __restrict__ Y,
                                                   int R, int Cp, int O) {
  __shared__ __align__(64) _Float16 tileB[2][64 * 32];   // 2 x 4KB
  const int wv   = threadIdx.x >> 5;
  const int lane = threadIdx.x & 31;
  const int l16  = lane & 15;
  const int hl   = lane >> 4;
  const int m0   = blockIdx.x * 64 + (wv & 3) * 16;
  const int nblk = blockIdx.y * 64;
  const int nl0  = (wv >> 2) * 32;                       // wave's local N base (0 or 32)
  const _Float16* xa = X + (size_t)(m0 + l16) * Cp + hl * 8;
  const _Float16* wrow = W + (size_t)nblk * Cp;
  const bool issuer = (wv == 0);

  v8f acc0 = {}, acc1 = {};
  if (issuer)
    tdm_load_b_tile(wrow, (unsigned)(uintptr_t)&tileB[0][0], Cp);

  const int nk = Cp >> 5;
  for (int t = 0; t < nk; ++t) {
    const int k0 = t << 5;
    const int cur = t & 1;
    if (issuer) __builtin_amdgcn_s_wait_tensorcnt(0);
    __syncthreads();                                     // tile[cur] ready; tile[cur^1] free
    if (issuer && (t + 1 < nk))
      tdm_load_b_tile(wrow + k0 + 32, (unsigned)(uintptr_t)&tileB[cur ^ 1][0], Cp);

    __builtin_prefetch(xa + k0 + 32, 0, 3);
    v8h a0 = *(const v8h*)(xa + k0);
    v8h a1 = *(const v8h*)(xa + k0 + 16);
    v16h a = __builtin_shufflevector(a0, a1, 0,1,2,3,4,5,6,7,8,9,10,11,12,13,14,15);

    const _Float16* base = &tileB[cur][0];
    const _Float16* pb0 = base + (nl0 + l16) * 32 + hl * 8;
    const _Float16* pb1 = base + (nl0 + 16 + l16) * 32 + hl * 8;
    v8h b0a = *(const v8h*)(pb0);
    v8h b0b = *(const v8h*)(pb0 + 16);
    v8h b1a = *(const v8h*)(pb1);
    v8h b1b = *(const v8h*)(pb1 + 16);
    v16h b0 = __builtin_shufflevector(b0a, b0b, 0,1,2,3,4,5,6,7,8,9,10,11,12,13,14,15);
    v16h b1 = __builtin_shufflevector(b1a, b1b, 0,1,2,3,4,5,6,7,8,9,10,11,12,13,14,15);

    acc0 = __builtin_amdgcn_wmma_f32_16x16x32_f16(false, a, false, b0, (short)0, acc0, false, false);
    acc1 = __builtin_amdgcn_wmma_f32_16x16x32_f16(false, a, false, b1, (short)0, acc1, false, false);
  }

  // D layout: VGPR v, lane -> row = m0 + v + hl*8, col = n + l16
  const int c0 = nblk + nl0 + l16;
  const float bv0 = bias[c0];
  const float bv1 = bias[c0 + 16];
  _Float16* y0 = Y + (size_t)(m0 + hl * 8) * O + c0;
#pragma unroll
  for (int v = 0; v < 8; ++v) {
    y0[(size_t)v * O]      = (_Float16)(acc0[v] + bv0);
    y0[(size_t)v * O + 16] = (_Float16)(acc1[v] + bv1);
  }
}

// ---------------------------------------------------------------------------
// Training-mode BN stats, coalesced: 32 consecutive channels per block,
// lanes -> consecutive channels, 8 row-groups reduced through LDS.
// ---------------------------------------------------------------------------
__global__ __launch_bounds__(256) void k_bn_stats(const _Float16* __restrict__ Y,
                                                  int Rows, int O,
                                                  float* __restrict__ mean,
                                                  float* __restrict__ rstd) {
  int cg = blockIdx.x * 32;
  int c  = threadIdx.x & 31;
  int rg = threadIdx.x >> 5;
  float s = 0.f, q = 0.f;
  for (int r = rg; r < Rows; r += 8) {
    float v = (float)Y[(size_t)r * O + cg + c];
    s += v; q += v * v;
  }
  __shared__ float ss[8][32], sq[8][32];
  ss[rg][c] = s; sq[rg][c] = q;
  __syncthreads();
  if (rg == 0) {
#pragma unroll
    for (int i = 1; i < 8; ++i) { s += ss[i][c]; q += sq[i][c]; }
    float m = s / (float)Rows;
    float var = q / (float)Rows - m * m;
    mean[cg + c] = m;
    rstd[cg + c] = rsqrtf(fmaxf(var, 0.f) + 1e-5f);
  }
}

__global__ void k_bn_relu(const _Float16* __restrict__ Y, _Float16* __restrict__ Xn,
                          int Rows, int O,
                          const float* __restrict__ mean, const float* __restrict__ rstd,
                          const float* __restrict__ g, const float* __restrict__ bt) {
  size_t i = (size_t)blockIdx.x * blockDim.x + threadIdx.x;
  size_t total = (size_t)Rows * O;
  if (i >= total) return;
  int c = (int)(i % O);
  float v = ((float)Y[i] - mean[c]) * rstd[c] * g[c] + bt[c];
  Xn[i] = (_Float16)fmaxf(v, 0.f);
}

// Maxpool over the nsample axis: one block per (b,s).
__global__ void k_maxpool(const _Float16* __restrict__ H, _Float16* __restrict__ out,
                          int ns, int O) {
  int bs = blockIdx.x;
  const _Float16* p = H + (size_t)bs * ns * O;
  for (int c = threadIdx.x; c < O; c += blockDim.x) {
    float m = -1e30f;
    for (int j = 0; j < ns; ++j) m = fmaxf(m, (float)p[(size_t)j * O + c]);
    out[(size_t)bs * O + c] = (_Float16)m;
  }
}

// Final heads: q = normalize(x@qw^T + qb), t = x@tw^T + tb.  One block per row.
__global__ __launch_bounds__(256) void k_head(const _Float16* __restrict__ x,
                                              const float* __restrict__ qw, const float* __restrict__ qb,
                                              const float* __restrict__ tw, const float* __restrict__ tb,
                                              float* __restrict__ out, int K, int B) {
  int b = blockIdx.x, t = threadIdx.x;
  __shared__ float red[256];
  __shared__ float o7[7];
  for (int o = 0; o < 7; ++o) {
    const float* w = (o < 4) ? (qw + (size_t)o * K) : (tw + (size_t)(o - 4) * K);
    float s = 0.f;
    for (int k = t; k < K; k += blockDim.x) s += (float)x[(size_t)b * K + k] * w[k];
    red[t] = s;
    __syncthreads();
    for (int st = 128; st > 0; st >>= 1) {
      if (t < st) red[t] += red[t + st];
      __syncthreads();
    }
    if (t == 0) o7[o] = red[0] + ((o < 4) ? qb[o] : tb[o - 4]);
    __syncthreads();
  }
  if (t == 0) {
    float q0 = o7[0], q1 = o7[1], q2 = o7[2], q3 = o7[3];
    float nrm = fmaxf(sqrtf(q0 * q0 + q1 * q1 + q2 * q2 + q3 * q3), 1e-12f);
    out[b * 4 + 0] = q0 / nrm;
    out[b * 4 + 1] = q1 / nrm;
    out[b * 4 + 2] = q2 / nrm;
    out[b * 4 + 3] = q3 / nrm;
    float* to = out + B * 4;
    to[b * 3 + 0] = o7[4];
    to[b * 3 + 1] = o7[5];
    to[b * 3 + 2] = o7[6];
  }
}

// ---------------------------------------------------------------------------
// Host side
// ---------------------------------------------------------------------------
static void run_layer(const _Float16* Xin, const _Float16* Wh, const float* bias,
                      _Float16* Ybuf, _Float16* Xout, int R, int Cp, int O,
                      const float* g, const float* bt, float* mean, float* rstd,
                      hipStream_t stream) {
  dim3 grid(R / 64, O / 64);
  k_gemm_wmma<<<grid, 256, 0, stream>>>(Xin, Wh, bias, Ybuf, R, Cp, O);
  k_bn_stats<<<O / 32, 256, 0, stream>>>(Ybuf, R, O, mean, rstd);
  size_t tot = (size_t)R * O;
  k_bn_relu<<<(unsigned)((tot + 255) / 256), 256, 0, stream>>>(Ybuf, Xout, R, O, mean, rstd, g, bt);
}

extern "C" void kernel_launch(void* const* d_in, const int* in_sizes, int n_in,
                              void* d_out, int out_size, void* d_ws, size_t ws_size,
                              hipStream_t stream) {
  (void)in_sizes; (void)n_in; (void)out_size; (void)ws_size;
  const float* pc = (const float*)d_in[0];   // [64,2048,3]
  const float* z  = (const float*)d_in[1];   // [64,2]
  auto P = [&](int i) { return (const float*)d_in[i]; };

  char* ws = (char*)d_ws;
  size_t off = 0;
  auto take = [&](size_t bytes) {
    void* p = ws + off;
    off = (off + bytes + 255) & ~(size_t)255;
    return p;
  };

  float*    mean  = (float*)take(1024 * 4);
  float*    rstd  = (float*)take(1024 * 4);
  float*    nx0   = (float*)take((size_t)64 * 128 * 3 * 4);   // new_xyz SA0
  float*    nx1   = (float*)take((size_t)64 * 32 * 3 * 4);    // new_xyz SA1
  _Float16* pts1  = (_Float16*)take((size_t)64 * 128 * 128 * 2);
  _Float16* pts2  = (_Float16*)take((size_t)64 * 32 * 256 * 2);
  _Float16* pool3 = (_Float16*)take((size_t)64 * 512 * 2);

  // Converted weights (f16, K padded to mult of 32): {d_in idx, O, C, Cp}
  struct WD { int idx, O, C, Cp; };
  const WD wd[11] = {
    {2, 64, 8, 32},    {6, 64, 64, 64},    {10, 128, 64, 64},
    {14, 128, 131, 160}, {18, 128, 128, 128}, {22, 256, 128, 128},
    {26, 256, 259, 288}, {30, 256, 256, 256}, {34, 512, 256, 256},
    {38, 1024, 512, 512}, {40, 1024, 1024, 1024}};
  _Float16* wh[11];
  for (int i = 0; i < 11; ++i) {
    int tot = wd[i].O * wd[i].Cp;
    wh[i] = (_Float16*)take((size_t)tot * 2);
    k_convert_w<<<(tot + 255) / 256, 256, 0, stream>>>(P(wd[i].idx), wh[i],
                                                       wd[i].O, wd[i].C, wd[i].Cp);
  }

  // Ping-pong activation buffers (max R*Cp = 524288*128 halves)
  _Float16* bufA = (_Float16*)take((size_t)67108864 * 2);
  _Float16* bufB = (_Float16*)take((size_t)67108864 * 2);

  // ---- SA0: npoint=128, radius=0.2, nsample=64, mlp 8->64->64->128 ----
  const int R0 = 64 * 128 * 64;
  k_fps<<<64, 256, 0, stream>>>(pc, 2048, 128, nx0);
  k_group0<<<64 * 128, 32, 0, stream>>>(pc, z, nx0, 0.04f, 2048, 128, 64, bufA, 32);
  run_layer(bufA, wh[0], P(3),  bufB, bufA, R0, 32, 64,  P(4),  P(5),  mean, rstd, stream);
  run_layer(bufA, wh[1], P(7),  bufB, bufA, R0, 64, 64,  P(8),  P(9),  mean, rstd, stream);
  run_layer(bufA, wh[2], P(11), bufB, bufA, R0, 64, 128, P(12), P(13), mean, rstd, stream);
  k_maxpool<<<64 * 128, 128, 0, stream>>>(bufA, pts1, 64, 128);

  // ---- SA1: npoint=32, radius=0.4, nsample=128, mlp 131->128->128->256 ----
  const int R1 = 64 * 32 * 128;
  k_fps<<<64, 256, 0, stream>>>(nx0, 128, 32, nx1);
  k_group1<<<64 * 32, 32, 0, stream>>>(nx0, pts1, nx1, 0.16f, 128, 32, 128, 128, bufA, 160);
  run_layer(bufA, wh[3], P(15), bufB, bufA, R1, 160, 128, P(16), P(17), mean, rstd, stream);
  run_layer(bufA, wh[4], P(19), bufB, bufA, R1, 128, 128, P(20), P(21), mean, rstd, stream);
  run_layer(bufA, wh[5], P(23), bufB, bufA, R1, 128, 256, P(24), P(25), mean, rstd, stream);
  k_maxpool<<<64 * 32, 256, 0, stream>>>(bufA, pts2, 128, 256);

  // ---- SA2 (group_all): mlp 259->256->256->512 over 32 points ----
  const int R2 = 64 * 32;
  {
    int tot = R2 * 288;
    k_group_all<<<(tot + 255) / 256, 256, 0, stream>>>(nx1, pts2, bufA, 64, 32, 256, 288);
  }
  run_layer(bufA, wh[6], P(27), bufB, bufA, R2, 288, 256, P(28), P(29), mean, rstd, stream);
  run_layer(bufA, wh[7], P(31), bufB, bufA, R2, 256, 256, P(32), P(33), mean, rstd, stream);
  run_layer(bufA, wh[8], P(35), bufB, bufA, R2, 256, 512, P(36), P(37), mean, rstd, stream);
  k_maxpool<<<64, 256, 0, stream>>>(bufA, pool3, 32, 512);

  // ---- FC head: 512->1024 (bn1) -> 1024->1024 (bn2) -> q/t ----
  run_layer(pool3, wh[9],  P(39), bufB, bufA, 64, 512,  1024, P(46), P(47), mean, rstd, stream);
  run_layer(bufA,  wh[10], P(41), bufB, bufA, 64, 1024, 1024, P(48), P(49), mean, rstd, stream);
  k_head<<<64, 256, 0, stream>>>(bufA, P(42), P(43), P(44), P(45), (float*)d_out, 1024, 64);
}
